// ContinuousThoughtMachine_15685220565451
// MI455X (gfx1250) — matrix operand order repaired and checked by reading, not compiled, not run on hardware
//
#include <hip/hip_runtime.h>

// ---------------------------------------------------------------------------
// CDNA5 (gfx1250) Continuous Thought Machine forward pass.
// Strategy: bf16-convert + transpose all weights once (fits in 192MB L2),
// then run all GEMMs through v_wmma_f32_16x16x32_bf16 (wave32, 16x16 tiles).
// ---------------------------------------------------------------------------

typedef __attribute__((ext_vector_type(8)))  __bf16 v8bf;
typedef __attribute__((ext_vector_type(16))) __bf16 v16bf;
typedef __attribute__((ext_vector_type(8)))  float  v8f;

union V16u { v16bf v; v8bf h[2]; };

__device__ __forceinline__ float gelu_f(float x) {
  return 0.5f * x * (1.0f + erff(x * 0.70710678118654752f));
}
__device__ __forceinline__ float softplus_f(float x) {
  return (x > 20.0f) ? x : log1pf(expf(x));
}

// ---------------------------------------------------------------------------
// Weight convert: fp32 W[K][N] -> bf16 Wt[N][K]  (LDS-tiled transpose)
// K, N multiples of 32.
// ---------------------------------------------------------------------------
__global__ void k_convT(const float* __restrict__ W, __bf16* __restrict__ Wt,
                        int K, int N) {
  __shared__ float tile[32][33];
  int tx = threadIdx.x & 31;
  int ty = threadIdx.x >> 5;           // 0..7
  int k0 = blockIdx.x * 32;
  int n0 = blockIdx.y * 32;
#pragma unroll
  for (int i = 0; i < 4; ++i)
    tile[ty + 8 * i][tx] = W[(size_t)(k0 + ty + 8 * i) * N + (n0 + tx)];
  __syncthreads();
#pragma unroll
  for (int i = 0; i < 4; ++i)
    Wt[(size_t)(n0 + ty + 8 * i) * K + (k0 + tx)] = (__bf16)tile[tx][ty + 8 * i];
}

// fp32 -> bf16 straight copy (ctx)
__global__ void k_conv(const float* __restrict__ in, __bf16* __restrict__ out,
                       size_t n) {
  size_t i = (size_t)blockIdx.x * blockDim.x + threadIdx.x;
  size_t stride = (size_t)gridDim.x * blockDim.x;
  for (; i < n; i += stride) out[i] = (__bf16)in[i];
}

// ---------------------------------------------------------------------------
// WMMA GEMM: C[M,N] = concat(A1[M,K1], A2[M,K2]) @ Wt[N,K]^T (+bias)
// A row-major bf16, Wt is [N][K] bf16 (pre-transposed). One wave per 16x16
// C tile. K multiples of 32, N multiples of 16, M multiples of 16.
// A-operand layout (16-bit A 16x32): lanes 0-15 row=lane, K base 0;
// lanes 16-31 row=lane-16, K base 8; VGPR4-7 hold K+16.
// B-operand layout (16-bit B 32x16): lane&15 = N col, same K order as A.
// ---------------------------------------------------------------------------
__global__ __launch_bounds__(32) void k_gemm(
    const __bf16* __restrict__ A1, int K1,
    const __bf16* __restrict__ A2, int K2,
    const __bf16* __restrict__ Wt,
    const float*  __restrict__ bias,
    float*  __restrict__ Cf, int ldCf,
    __bf16* __restrict__ Cb, int ldCb) {
  const int K    = K1 + K2;
  const int lane = threadIdx.x & 31;
  const int row0 = blockIdx.y << 4;
  const int col0 = blockIdx.x << 4;
  const int rl   = lane & 15;
  const int hsel = lane >> 4;
  const int kbA  = hsel * 8;    // A half-select
  const int kbB  = hsel * 8;    // B half-select

  const __bf16* wrow = Wt + (size_t)(col0 + rl) * K;
  v8f acc = {};

  const __bf16* arow1 = A1 + (size_t)(row0 + rl) * K1;
  for (int kc = 0; kc < K1; kc += 32) {
    V16u a, b;
    a.h[0] = *(const v8bf*)(arow1 + kc + kbA);
    a.h[1] = *(const v8bf*)(arow1 + kc + 16 + kbA);
    const __bf16* bp = wrow + kc + kbB;
    b.h[0] = *(const v8bf*)(bp);
    b.h[1] = *(const v8bf*)(bp + 16);
    __builtin_prefetch(bp + 256, 0, 1);  // global_prefetch_b8 next weight chunk
    acc = __builtin_amdgcn_wmma_f32_16x16x32_bf16(
        false, a.v, false, b.v, (short)0, acc, false, false);
  }
  if (A2) {
    const __bf16* arow2 = A2 + (size_t)(row0 + rl) * K2;
    for (int kc = 0; kc < K2; kc += 32) {
      V16u a, b;
      a.h[0] = *(const v8bf*)(arow2 + kc + kbA);
      a.h[1] = *(const v8bf*)(arow2 + kc + 16 + kbA);
      const __bf16* bp = wrow + K1 + kc + kbB;
      b.h[0] = *(const v8bf*)(bp);
      b.h[1] = *(const v8bf*)(bp + 16);
      acc = __builtin_amdgcn_wmma_f32_16x16x32_bf16(
          false, a.v, false, b.v, (short)0, acc, false, false);
    }
  }
#pragma unroll
  for (int i = 0; i < 8; ++i) {
    int row = row0 + i + 8 * hsel;
    int col = col0 + rl;
    float v = acc[i];
    if (bias) v += bias[col];
    if (Cf) Cf[(size_t)row * ldCf + col] = v;
    if (Cb) Cb[(size_t)row * ldCb + col] = (__bf16)v;
  }
}

// ---------------------------------------------------------------------------
// LayerNorm (+ optional exact GELU). One block per row; 16 rows per tensor.
// ---------------------------------------------------------------------------
__global__ void k_ln(const float* __restrict__ X, int N,
                     const float* __restrict__ g, const float* __restrict__ b,
                     __bf16* __restrict__ outB, float* __restrict__ outF,
                     int doGelu) {
  int row = blockIdx.x;
  const float* x = X + (size_t)row * N;
  __shared__ float red[256];
  int tid = threadIdx.x;
  float s = 0.f;
  for (int j = tid; j < N; j += 256) s += x[j];
  red[tid] = s; __syncthreads();
  for (int off = 128; off > 0; off >>= 1) {
    if (tid < off) red[tid] += red[tid + off];
    __syncthreads();
  }
  float mu = red[0] / (float)N; __syncthreads();
  float vs = 0.f;
  for (int j = tid; j < N; j += 256) { float d = x[j] - mu; vs += d * d; }
  red[tid] = vs; __syncthreads();
  for (int off = 128; off > 0; off >>= 1) {
    if (tid < off) red[tid] += red[tid + off];
    __syncthreads();
  }
  float inv = rsqrtf(red[0] / (float)N + 1e-5f);
  for (int j = tid; j < N; j += 256) {
    float y = (x[j] - mu) * inv * g[j] + b[j];
    if (doGelu) y = gelu_f(y);
    if (outB) outB[(size_t)row * N + j] = (__bf16)y;
    if (outF) outF[(size_t)row * N + j] = y;
  }
}

// ---------------------------------------------------------------------------
// Incremental synchronization: num_t = num_{t-1}*e^{-r} + z_i*z_j,
// den_t = geometric sum of decay (closed form). tlen = history length.
// ---------------------------------------------------------------------------
__global__ void k_sync(const float* __restrict__ z,
                       const float* __restrict__ act_dec,
                       const int* __restrict__ ai, const int* __restrict__ aj,
                       const float* __restrict__ out_dec,
                       const int* __restrict__ oi, const int* __restrict__ oj,
                       float* __restrict__ accA, float* __restrict__ accO,
                       __bf16* __restrict__ sactB, float* __restrict__ soutF,
                       int tlen) {
  int p = blockIdx.x * 256 + threadIdx.x;   // 0..1023
  int b = blockIdx.y;
  bool isAct = p < 512;
  int pp = isAct ? p : p - 512;
  const float* dec = isAct ? act_dec : out_dec;
  const int* pi = isAct ? ai : oi;
  const int* pj = isAct ? aj : oj;
  float r = softplus_f(dec[pp]);
  float e = expf(-r);
  float* acc = (isAct ? accA : accO) + b * 512 + pp;
  float num = acc[0] * e + z[b * 1024 + pi[pp]] * z[b * 1024 + pj[pp]];
  acc[0] = num;
  float ome = 1.f - e;
  float den = (ome > 1e-6f) ? (1.f - expf(-r * (float)tlen)) / ome : (float)tlen;
  float sv = num * rsqrtf(den + 1e-8f);
  if (isAct) {
    sactB[b * 512 + pp] = (__bf16)sv;
    if (b == 0)
      for (int rr = 8; rr < 16; ++rr) sactB[rr * 512 + pp] = (__bf16)0.f;
  } else {
    soutF[b * 512 + pp] = sv;
  }
}

// ---------------------------------------------------------------------------
// Attention: scores + softmax over S=256. Block = (h,b), thread = s.
// ---------------------------------------------------------------------------
__global__ void k_scores(const float* __restrict__ qh,
                         const float* __restrict__ kbuf,
                         float* __restrict__ wbuf) {
  int h = blockIdx.x, b = blockIdx.y, s = threadIdx.x;
  const float* qp = qh + b * 1024 + h * 128;
  const float* kp = kbuf + ((size_t)(b * 256 + s)) * 1024 + h * 128;
  float acc = 0.f;
#pragma unroll 4
  for (int d = 0; d < 128; ++d) acc = fmaf(qp[d], kp[d], acc);
  acc *= 0.088388347648318447f;  // 1/sqrt(128)
  __shared__ float red[256];
  red[s] = acc; __syncthreads();
  for (int off = 128; off > 0; off >>= 1) {
    if (s < off) red[s] = fmaxf(red[s], red[s + off]);
    __syncthreads();
  }
  float mx = red[0]; __syncthreads();
  float ex = expf(acc - mx);
  red[s] = ex; __syncthreads();
  for (int off = 128; off > 0; off >>= 1) {
    if (s < off) red[s] += red[s + off];
    __syncthreads();
  }
  wbuf[((size_t)(b * 8 + h)) * 256 + s] = ex / red[0];
}

__global__ void k_attnmean(const float* __restrict__ wbuf,
                           float* __restrict__ out, int t) {
  int b = blockIdx.x, s = threadIdx.x;
  float acc = 0.f;
  for (int h = 0; h < 8; ++h) acc += wbuf[(b * 8 + h) * 256 + s];
  out[200 + ((size_t)b * 12 + t) * 256 + s] = acc * 0.125f;
}

__global__ void k_oeinsum(const float* __restrict__ wbuf,
                          const float* __restrict__ vbuf,
                          __bf16* __restrict__ obuf) {
  int h = blockIdx.x, b = blockIdx.y, d = threadIdx.x;  // 128 threads
  const float* wp = wbuf + (b * 8 + h) * 256;
  const float* vp = vbuf + ((size_t)b * 256) * 1024 + h * 128 + d;
  float acc = 0.f;
  for (int s = 0; s < 256; ++s) acc = fmaf(wp[s], vp[(size_t)s * 1024], acc);
  obuf[b * 1024 + h * 128 + d] = (__bf16)acc;
  if (b == 0)
    for (int rr = 8; rr < 16; ++rr) obuf[rr * 1024 + h * 128 + d] = (__bf16)0.f;
}

// ---------------------------------------------------------------------------
// Neuron-level models: shift hist, hz = gelu(hist @ w0 + b0), z = hz@w1 + b1.
// One block per d (1024 blocks), 64 threads = hidden dim.
// ---------------------------------------------------------------------------
__global__ __launch_bounds__(64) void k_nlm(
    const float* __restrict__ w0, const float* __restrict__ b0,
    const float* __restrict__ w1, const float* __restrict__ b1,
    float* __restrict__ hist, const float* __restrict__ abuf,
    float* __restrict__ zbuf, __bf16* __restrict__ x0in) {
  int d = blockIdx.x, h = threadIdx.x;
  __shared__ float W0[25 * 64];
  __shared__ float hs[25];
  __shared__ float red[64];
  for (int m = 0; m < 25; ++m) W0[m * 64 + h] = w0[((size_t)d * 25 + m) * 64 + h];
  float w1v = w1[d * 64 + h];
  float b0v = b0[d * 64 + h];
  __syncthreads();
  for (int b = 0; b < 8; ++b) {
    float* hp = hist + ((size_t)(b * 1024 + d)) * 25;
    float av = abuf[b * 1024 + d];
    if (h < 25) hs[h] = (h < 24) ? hp[h + 1] : av;
    __syncthreads();
    float acc = b0v;
#pragma unroll
    for (int m = 0; m < 25; ++m) acc = fmaf(hs[m], W0[m * 64 + h], acc);
    red[h] = gelu_f(acc) * w1v;
    __syncthreads();
    for (int off = 32; off > 0; off >>= 1) {
      if (h < off) red[h] += red[h + off];
      __syncthreads();
    }
    if (h == 0) {
      float zv = red[0] + b1[d];
      zbuf[b * 1024 + d] = zv;
      x0in[b * 2048 + d] = (__bf16)zv;
    }
    if (h < 25) hp[h] = hs[h];  // write back shifted history
    __syncthreads();
  }
}

// ---------------------------------------------------------------------------
// Prediction head: p = sigmoid(LN(gelu(s_out@h0w+h0b)) @ h1w + h1b)
// ---------------------------------------------------------------------------
__global__ void k_head(const float* __restrict__ soutF,
                       const float* __restrict__ h0w, const float* __restrict__ h0b,
                       const float* __restrict__ hg, const float* __restrict__ hb,
                       const float* __restrict__ h1w, const float* __restrict__ h1b,
                       float* __restrict__ out, int t) {
  int b = blockIdx.x, j = threadIdx.x;  // 256 threads
  __shared__ float red[256];
  const float* sp = soutF + b * 512;
  float acc = h0b[j];
  for (int i = 0; i < 512; ++i) acc = fmaf(sp[i], h0w[i * 256 + j], acc);
  float g = gelu_f(acc);
  red[j] = g; __syncthreads();
  for (int off = 128; off > 0; off >>= 1) { if (j < off) red[j] += red[j + off]; __syncthreads(); }
  float mu = red[0] * (1.f / 256.f); __syncthreads();
  float dv = g - mu;
  red[j] = dv * dv; __syncthreads();
  for (int off = 128; off > 0; off >>= 1) { if (j < off) red[j] += red[j + off]; __syncthreads(); }
  float inv = rsqrtf(red[0] * (1.f / 256.f) + 1e-5f); __syncthreads();
  float y = (g - mu) * inv * hg[j] + hb[j];
  red[j] = y * h1w[j]; __syncthreads();
  for (int off = 128; off > 0; off >>= 1) { if (j < off) red[j] += red[j + off]; __syncthreads(); }
  if (j == 0) {
    float p = red[0] + h1b[0];
    out[b * 12 + t] = 1.f / (1.f + expf(-p));
  }
}

// ---------------------------------------------------------------------------
// State init: z broadcast, x0in z-part + zero pad rows, hist broadcast,
// sync accumulators zero. Re-run every call -> deterministic graph replay.
// ---------------------------------------------------------------------------
__global__ void k_init(const float* __restrict__ z_init,
                       const float* __restrict__ hist_init,
                       float* __restrict__ zbuf, __bf16* __restrict__ x0in,
                       float* __restrict__ hist,
                       float* __restrict__ accA, float* __restrict__ accO) {
  size_t gid = (size_t)blockIdx.x * 256 + threadIdx.x;
  size_t stride = (size_t)gridDim.x * 256;
  for (size_t i = gid; i < 8192; i += stride) {
    int dd = (int)(i & 1023);
    float zv = z_init[dd];
    zbuf[i] = zv;
    x0in[(i >> 10) * 2048 + dd] = (__bf16)zv;
  }
  for (size_t i = gid; i < 8 * 2048; i += stride)
    x0in[8 * 2048 + i] = (__bf16)0.f;        // pad rows 8..15 (both halves)
  for (size_t i = gid; i < 204800; i += stride)
    hist[i] = hist_init[i % 25600];
  for (size_t i = gid; i < 4096; i += stride) { accA[i] = 0.f; accO[i] = 0.f; }
}

// Entropy -> certainty -> argmax -> final.
__global__ void k_final(float* __restrict__ out) {
  __shared__ float certs[96];
  int tid = threadIdx.x;  // 128 threads
  if (tid < 96) {
    float p = out[tid];
    float ent = -(p * log2f(p + 1e-8f) + (1.f - p) * log2f(1.f - p + 1e-8f));
    float c = 1.f - ent;
    out[96 + tid] = c;
    certs[tid] = c;
  }
  __syncthreads();
  if (tid < 8) {
    int best = 0; float bv = certs[tid * 12];
    for (int t = 1; t < 12; ++t) {
      float cv = certs[tid * 12 + t];
      if (cv > bv) { bv = cv; best = t; }
    }
    out[192 + tid] = out[tid * 12 + best];
  }
}

// ---------------------------------------------------------------------------
extern "C" void kernel_launch(void* const* d_in, const int* in_sizes, int n_in,
                              void* d_out, int out_size, void* d_ws, size_t ws_size,
                              hipStream_t stream) {
  (void)in_sizes; (void)n_in; (void)out_size; (void)ws_size;
  const float* ctx       = (const float*)d_in[0];
  const float* z_init    = (const float*)d_in[1];
  const float* hist_init = (const float*)d_in[2];
  const float* nlm_w0    = (const float*)d_in[3];
  const float* nlm_b0    = (const float*)d_in[4];
  const float* nlm_w1    = (const float*)d_in[5];
  const float* nlm_b1    = (const float*)d_in[6];
  const float* syn_in_w  = (const float*)d_in[7];
  const float* syn_in_b  = (const float*)d_in[8];
  const float* enc0_w    = (const float*)d_in[9];
  const float* enc0_b    = (const float*)d_in[10];
  const float* enc0_g    = (const float*)d_in[11];
  const float* enc0_beta = (const float*)d_in[12];
  const float* enc1_w    = (const float*)d_in[13];
  const float* enc1_b    = (const float*)d_in[14];
  const float* enc1_g    = (const float*)d_in[15];
  const float* enc1_beta = (const float*)d_in[16];
  const float* bot_w     = (const float*)d_in[17];
  const float* bot_b     = (const float*)d_in[18];
  const float* bot_g     = (const float*)d_in[19];
  const float* bot_beta  = (const float*)d_in[20];
  const float* skip0_w   = (const float*)d_in[21];
  const float* skip0_b   = (const float*)d_in[22];
  const float* dec0_w    = (const float*)d_in[23];
  const float* dec0_b    = (const float*)d_in[24];
  const float* dec0_g    = (const float*)d_in[25];
  const float* dec0_beta = (const float*)d_in[26];
  const float* skip1_w   = (const float*)d_in[27];
  const float* skip1_b   = (const float*)d_in[28];
  const float* dec1_w    = (const float*)d_in[29];
  const float* dec1_b    = (const float*)d_in[30];
  const float* dec1_g    = (const float*)d_in[31];
  const float* dec1_beta = (const float*)d_in[32];
  const float* syn_out_w = (const float*)d_in[33];
  const float* syn_out_b = (const float*)d_in[34];
  const float* act_decay = (const float*)d_in[35];
  const float* out_decay = (const float*)d_in[36];
  const float* qproj_w   = (const float*)d_in[37];
  const float* qproj_b   = (const float*)d_in[38];
  const float* qln_g     = (const float*)d_in[39];
  const float* qln_b     = (const float*)d_in[40];
  const float* attn_wq   = (const float*)d_in[41];
  const float* attn_bq   = (const float*)d_in[42];
  const float* attn_wk   = (const float*)d_in[43];
  const float* attn_bk   = (const float*)d_in[44];
  const float* attn_wv   = (const float*)d_in[45];
  const float* attn_bv   = (const float*)d_in[46];
  const float* attn_wo   = (const float*)d_in[47];
  const float* attn_bo   = (const float*)d_in[48];
  const float* head0_w   = (const float*)d_in[49];
  const float* head0_b   = (const float*)d_in[50];
  const float* hln_g     = (const float*)d_in[51];
  const float* hln_b     = (const float*)d_in[52];
  const float* head1_w   = (const float*)d_in[53];
  const float* head1_b   = (const float*)d_in[54];
  const int*   act_i     = (const int*)d_in[55];
  const int*   act_j     = (const int*)d_in[56];
  const int*   out_i     = (const int*)d_in[57];
  const int*   out_j     = (const int*)d_in[58];
  float* out = (float*)d_out;

  // --- bump allocator over d_ws ---
  char* wp = (char*)d_ws;
  auto alloc = [&](size_t bytes) -> char* {
    char* p = wp; wp += (bytes + 255) & ~(size_t)255; return p;
  };
  auto allocB = [&](size_t n) { return (__bf16*)alloc(n * sizeof(__bf16)); };
  auto allocF = [&](size_t n) { return (float*)alloc(n * sizeof(float)); };

  // bf16 transposed weights [N][K] (total ~118 MB -> lives in 192MB L2)
  __bf16* synT   = allocB(1024ull * 2048);
  __bf16* enc0T  = allocB(2048ull * 1024);
  __bf16* enc1T  = allocB(4096ull * 2048);
  __bf16* botT   = allocB(4096ull * 4096);
  __bf16* skip0T = allocB(2048ull * 8192);
  __bf16* dec0T  = allocB(2048ull * 2048);
  __bf16* skip1T = allocB(1024ull * 4096);
  __bf16* dec1T  = allocB(1024ull * 1024);
  __bf16* soT    = allocB(1024ull * 1024);
  __bf16* qprojT = allocB(1024ull * 512);
  __bf16* wqT    = allocB(1024ull * 1024);
  __bf16* wkT    = allocB(1024ull * 1024);
  __bf16* wvT    = allocB(1024ull * 1024);
  __bf16* woT    = allocB(1024ull * 1024);
  __bf16* ctxb   = allocB(2048ull * 1024);

  // fp32 state / intermediates
  float* kbuf  = allocF(2048ull * 1024);
  float* vbuf  = allocF(2048ull * 1024);
  float* tmpC  = allocF(16ull * 4096);
  float* qh    = allocF(16ull * 1024);
  float* wbuf  = allocF(8ull * 8 * 256);
  float* zbuf  = allocF(8ull * 1024);
  float* abuf  = allocF(16ull * 1024);
  float* hist  = allocF(8ull * 1024 * 25);
  float* accA  = allocF(8ull * 512);
  float* accO  = allocF(8ull * 512);
  float* soutF = allocF(8ull * 512);

  // bf16 activations (16 rows: batch 8 padded to a WMMA M tile)
  __bf16* x0in  = allocB(16ull * 2048);  // [z | o]
  __bf16* x0buf = allocB(16ull * 1024);
  __bf16* e0    = allocB(16ull * 2048);
  __bf16* e1    = allocB(16ull * 4096);
  __bf16* xb    = allocB(16ull * 4096);
  __bf16* s0    = allocB(16ull * 2048);
  __bf16* d0    = allocB(16ull * 2048);
  __bf16* s1    = allocB(16ull * 1024);
  __bf16* d1    = allocB(16ull * 1024);
  __bf16* qbf   = allocB(16ull * 1024);
  __bf16* sact  = allocB(16ull * 512);
  __bf16* obuf  = allocB(16ull * 1024);

  auto convT = [&](const float* W, __bf16* Wt, int K, int N) {
    k_convT<<<dim3(K / 32, N / 32), 256, 0, stream>>>(W, Wt, K, N);
  };
  auto gemm = [&](const __bf16* A1, int K1, const __bf16* A2, int K2,
                  const __bf16* Wt, const float* bias, int N, int Mtiles,
                  float* Cf, int ldCf, __bf16* Cb, int ldCb) {
    k_gemm<<<dim3(N / 16, Mtiles), 32, 0, stream>>>(A1, K1, A2, K2, Wt, bias,
                                                    Cf, ldCf, Cb, ldCb);
  };

  // ---- one-time (per call) weight conversion + transpose ----
  convT(syn_in_w,  synT,   2048, 1024);
  convT(enc0_w,    enc0T,  1024, 2048);
  convT(enc1_w,    enc1T,  2048, 4096);
  convT(bot_w,     botT,   4096, 4096);
  convT(skip0_w,   skip0T, 8192, 2048);
  convT(dec0_w,    dec0T,  2048, 2048);
  convT(skip1_w,   skip1T, 4096, 1024);
  convT(dec1_w,    dec1T,  1024, 1024);
  convT(syn_out_w, soT,    1024, 1024);
  convT(qproj_w,   qprojT,  512, 1024);
  convT(attn_wq,   wqT,    1024, 1024);
  convT(attn_wk,   wkT,    1024, 1024);
  convT(attn_wv,   wvT,    1024, 1024);
  convT(attn_wo,   woT,    1024, 1024);
  k_conv<<<512, 256, 0, stream>>>(ctx, ctxb, 2048ull * 1024);

  // ---- K/V projections over full context (M = B*S = 2048) ----
  gemm(ctxb, 1024, nullptr, 0, wkT, attn_bk, 1024, 128, kbuf, 1024, nullptr, 0);
  gemm(ctxb, 1024, nullptr, 0, wvT, attn_bv, 1024, 128, vbuf, 1024, nullptr, 0);

  // ---- recurrent state init ----
  k_init<<<256, 256, 0, stream>>>(z_init, hist_init, zbuf, x0in, hist, accA, accO);

  // ---- T = 12 thought steps ----
  for (int t = 0; t < 12; ++t) {
    k_sync<<<dim3(4, 8), 256, 0, stream>>>(zbuf, act_decay, act_i, act_j,
                                           out_decay, out_i, out_j,
                                           accA, accO, sact, soutF, t + 1);
    // q = LN(s_act @ qproj + b)
    gemm(sact, 512, nullptr, 0, qprojT, qproj_b, 1024, 1, tmpC, 1024, nullptr, 0);
    k_ln<<<16, 256, 0, stream>>>(tmpC, 1024, qln_g, qln_b, qbf, nullptr, 0);
    // qh = q @ wq + bq
    gemm(qbf, 1024, nullptr, 0, wqT, attn_bq, 1024, 1, qh, 1024, nullptr, 0);
    // attention
    k_scores<<<dim3(8, 8), 256, 0, stream>>>(qh, kbuf, wbuf);
    k_attnmean<<<8, 256, 0, stream>>>(wbuf, out, t);
    k_oeinsum<<<dim3(8, 8), 128, 0, stream>>>(wbuf, vbuf, obuf);
    // o @ wo + bo -> second half of x0in
    gemm(obuf, 1024, nullptr, 0, woT, attn_bo, 1024, 1, nullptr, 0, x0in + 1024, 2048);
    // U-net synapse stack
    gemm(x0in, 2048, nullptr, 0, synT, syn_in_b, 1024, 1, nullptr, 0, x0buf, 1024);
    gemm(x0buf, 1024, nullptr, 0, enc0T, enc0_b, 2048, 1, tmpC, 2048, nullptr, 0);
    k_ln<<<16, 256, 0, stream>>>(tmpC, 2048, enc0_g, enc0_beta, e0, nullptr, 1);
    gemm(e0, 2048, nullptr, 0, enc1T, enc1_b, 4096, 1, tmpC, 4096, nullptr, 0);
    k_ln<<<16, 256, 0, stream>>>(tmpC, 4096, enc1_g, enc1_beta, e1, nullptr, 1);
    gemm(e1, 4096, nullptr, 0, botT, bot_b, 4096, 1, tmpC, 4096, nullptr, 0);
    k_ln<<<16, 256, 0, stream>>>(tmpC, 4096, bot_g, bot_beta, xb, nullptr, 1);
    gemm(xb, 4096, e1, 4096, skip0T, skip0_b, 2048, 1, nullptr, 0, s0, 2048);
    gemm(s0, 2048, nullptr, 0, dec0T, dec0_b, 2048, 1, tmpC, 2048, nullptr, 0);
    k_ln<<<16, 256, 0, stream>>>(tmpC, 2048, dec0_g, dec0_beta, d0, nullptr, 1);
    gemm(d0, 2048, e0, 2048, skip1T, skip1_b, 1024, 1, nullptr, 0, s1, 1024);
    gemm(s1, 1024, nullptr, 0, dec1T, dec1_b, 1024, 1, tmpC, 1024, nullptr, 0);
    k_ln<<<16, 256, 0, stream>>>(tmpC, 1024, dec1_g, dec1_beta, d1, nullptr, 1);
    gemm(d1, 1024, nullptr, 0, soT, syn_out_b, 1024, 1, abuf, 1024, nullptr, 0);
    // neuron-level models: hist shift + z update (also refreshes x0in z-part)
    k_nlm<<<1024, 64, 0, stream>>>(nlm_w0, nlm_b0, nlm_w1, nlm_b1,
                                   hist, abuf, zbuf, x0in);
    // prediction head
    k_head<<<8, 256, 0, stream>>>(soutF, head0_w, head0_b, hln_g, hln_b,
                                  head1_w, head1_b, out, t);
  }

  // certainty, argmax, final
  k_final<<<1, 128, 0, stream>>>(out);
}
